// MaskedConv2d_49555332662086
// MI455X (gfx1250) — matrix operand order repaired and checked by reading, not compiled, not run on hardware
//
#include <hip/hip_runtime.h>

// MaskedConv2d inference for MI455X (gfx1250, wave32).
// Implicit-GEMM 3x3 conv: out[256,147456] = W[256,2304] x patches[2304,147456]
//   - v_wmma_f32_16x16x32_bf16, 4 WMMA / wave / K-step (128co x 64px block tile)
//   - weights pre-converted once to bf16 Wt[tap][co][ci] in d_ws -> A staging is
//     pure b128 copy (no per-iteration cvt), L2-resident (1.18 MB vs 192 MB L2)
//   - double-buffered LDS (As/Bs x2 = 30 KB), one barrier per K-step, next-step
//     global loads issued before the WMMAs so latency hides under matrix work
//   - fused gate = dense + sparse*spa_mask, ReLU epilogue
// Roofline: 174 GFLOP vs ~310 MB fp32 HBM traffic (~13 us @ 23.3 TB/s) ->
// compute-bound on the bf16 WMMA pipe (K=32/instr vs K=4 for f32 WMMA).

typedef __attribute__((ext_vector_type(16))) __bf16 v16bf;
typedef __attribute__((ext_vector_type(8)))  float  v8f;

#define H_IMG 384
#define W_IMG 384
#define C_CH  256
#define HW    (H_IMG * W_IMG)
#define LDSTRIDE 40   // halves per LDS row: 80 B -> every row 16B-aligned

__device__ __forceinline__ unsigned short f2bf_rne(float f) {
    unsigned int u = __float_as_uint(f);
    u += 0x7FFFu + ((u >> 16) & 1u);       // round-to-nearest-even
    return (unsigned short)(u >> 16);
}

union Frag16 { unsigned int u[8]; v16bf v; };

// ---- one-time weight transform: wgt[co][ci][tap] f32 -> Wt[tap][co][ci] bf16 ----
__global__ __launch_bounds__(256)
void prep_weights(const float* __restrict__ wgt, unsigned short* __restrict__ Wt)
{
    const int idx = blockIdx.x * 256 + threadIdx.x;   // 0 .. 9*256*256-1
    const int tap = idx >> 16;
    const int rem = idx & 0xFFFF;
    const int co  = rem >> 8;
    const int ci  = rem & 255;
    Wt[idx] = f2bf_rne(wgt[((size_t)co * C_CH + ci) * 9 + tap]);
}

__global__ __launch_bounds__(256)
void masked_conv3x3_wmma_bf16(const float* __restrict__ x,
                              const int*   __restrict__ spa,
                              const unsigned short* __restrict__ Wt,
                              const float* __restrict__ chm,
                              float*       __restrict__ out)
{
    __shared__ alignas(16) unsigned short As[2][128 * LDSTRIDE]; // [co_local][ci]  (M x K)
    __shared__ alignas(16) unsigned short Bs[2][64 * LDSTRIDE];  // [pix_local][ci] (N x K)

    const int tid   = threadIdx.x;
    const int lane  = tid & 31;
    const int wave  = tid >> 5;
    const int l16   = lane & 15;
    const int lhalf = lane >> 4;            // 0 or 1
    const int mi2   = (wave & 3) * 2;       // two 16-wide M tiles: mi2, mi2+1
    const int nt    = (wave >> 2) * 2;      // two 16-wide N tiles: nt, nt+1

    const int sblk = blockIdx.x;            // 0..2303: 384 rows x 6 segments
    const int hrow = sblk / (W_IMG / 64);
    const int w0   = (sblk % (W_IMG / 64)) * 64;
    const int coB  = blockIdx.y * 128;      // output-channel block base

    // cooperative-staging indices
    const int arow = tid & 127;             // A: co row, each thread 2x uint4
    const int aq   = tid >> 7;              // A: which 16-half group
    const int pl   = tid & 63;              // B: pixel row
    const int quad = tid >> 6;              // B: which K octet

    v8f acc00 = {}, acc01 = {}, acc10 = {}, acc11 = {};

    // ---- staging helpers ----
    auto loadA = [&](int it, uint4& r0, uint4& r1) {
        const int tap = it >> 3, c0 = (it & 7) << 5;
        const uint4* srcA = reinterpret_cast<const uint4*>(
            Wt + ((size_t)tap * C_CH + (coB + arow)) * C_CH + c0 + aq * 16);
        r0 = srcA[0];
        r1 = srcA[1];
    };
    auto storeA = [&](int buf, uint4 r0, uint4 r1) {
        *reinterpret_cast<uint4*>(&As[buf][arow * LDSTRIDE + aq * 16])     = r0;
        *reinterpret_cast<uint4*>(&As[buf][arow * LDSTRIDE + aq * 16 + 8]) = r1;
    };
    auto loadB = [&](int it, unsigned int (&pk)[4]) {
        const int tap = it >> 3, c0 = (it & 7) << 5;
        const int dy = tap / 3 - 1, dx = tap % 3 - 1;
        const int hh = hrow + dy, ww = w0 + pl + dx;
        const bool inb = (hh >= 0) & (hh < H_IMG) & (ww >= 0) & (ww < W_IMG);
        const float* xp = x + ((size_t)(c0 + quad * 8) * H_IMG + hh) * W_IMG + ww;
#pragma unroll
        for (int j = 0; j < 4; ++j) {
            float f0 = inb ? xp[(size_t)(2 * j    ) * HW] : 0.0f;
            float f1 = inb ? xp[(size_t)(2 * j + 1) * HW] : 0.0f;
            pk[j] = (unsigned int)f2bf_rne(f0) | ((unsigned int)f2bf_rne(f1) << 16);
        }
        if (inb && (c0 + 32) < C_CH)
            __builtin_prefetch(xp + (size_t)32 * HW, 0, 1);   // global_prefetch_b8
    };
    auto storeB = [&](int buf, const unsigned int (&pk)[4]) {
        *reinterpret_cast<uint4*>(&Bs[buf][pl * LDSTRIDE + quad * 8]) =
            make_uint4(pk[0], pk[1], pk[2], pk[3]);
    };

    // ---- fragment builders (documented wave32 VGPR layouts) ----
    // A 16x32: lane<16 -> K {0..7,16..23}; lane>=16 -> K {8..15,24..31}
    auto fragA = [&](int buf, int mtile) {
        Frag16 f;
        const int m = mtile * 16 + l16;
        const uint4 lo = *reinterpret_cast<const uint4*>(&As[buf][m * LDSTRIDE + lhalf * 8]);
        const uint4 hi = *reinterpret_cast<const uint4*>(&As[buf][m * LDSTRIDE + lhalf * 8 + 16]);
        f.u[0] = lo.x; f.u[1] = lo.y; f.u[2] = lo.z; f.u[3] = lo.w;
        f.u[4] = hi.x; f.u[5] = hi.y; f.u[6] = hi.z; f.u[7] = hi.w;
        return f;
    };
    // B 32x16: column = lane%16; lane<16 -> K {0..7,16..23}; lane>=16 -> K {8..15,24..31}
    auto fragB = [&](int buf, int ntile) {
        Frag16 f;
        const int n = ntile * 16 + l16;
        const uint4 lo = *reinterpret_cast<const uint4*>(&Bs[buf][n * LDSTRIDE + lhalf * 8]);
        const uint4 hi = *reinterpret_cast<const uint4*>(&Bs[buf][n * LDSTRIDE + lhalf * 8 + 16]);
        f.u[0] = lo.x; f.u[1] = lo.y; f.u[2] = lo.z; f.u[3] = lo.w;
        f.u[4] = hi.x; f.u[5] = hi.y; f.u[6] = hi.z; f.u[7] = hi.w;
        return f;
    };

    uint4 ra0, ra1;
    unsigned int pb[4];

    // prologue: stage K-step 0 into buffer 0
    loadA(0, ra0, ra1);
    loadB(0, pb);
    storeA(0, ra0, ra1);
    storeB(0, pb);
    __syncthreads();

    // K = 9 taps x 8 chunks of 32 input channels = 72 steps, double-buffered
    for (int it = 0; it < 72; ++it) {
        const int  cur  = it & 1;
        const bool more = (it + 1) < 72;

        // issue next step's global loads first; latency hides under the WMMAs
        if (more) { loadA(it + 1, ra0, ra1); loadB(it + 1, pb); }

        const Frag16 a0 = fragA(cur, mi2);
        const Frag16 a1 = fragA(cur, mi2 + 1);
        const Frag16 b0 = fragB(cur, nt);
        const Frag16 b1 = fragB(cur, nt + 1);

        acc00 = __builtin_amdgcn_wmma_f32_16x16x32_bf16(false, a0.v, false, b0.v, (short)0, acc00, false, false);
        acc01 = __builtin_amdgcn_wmma_f32_16x16x32_bf16(false, a0.v, false, b1.v, (short)0, acc01, false, false);
        acc10 = __builtin_amdgcn_wmma_f32_16x16x32_bf16(false, a1.v, false, b0.v, (short)0, acc10, false, false);
        acc11 = __builtin_amdgcn_wmma_f32_16x16x32_bf16(false, a1.v, false, b1.v, (short)0, acc11, false, false);

        if (more) { storeA(cur ^ 1, ra0, ra1); storeB(cur ^ 1, pb); }
        __syncthreads();
    }

    // ---- epilogue: gate = dense + sparse*spa, out = relu(y*gate) ----
    // C/D layout: VGPR r -> M = 8*(lane/16) + r, N = lane%16
    const int wc0 = w0 + nt * 16 + l16;
    const int wc1 = w0 + (nt + 1) * 16 + l16;
    const float s0 = (float)spa[hrow * W_IMG + wc0];
    const float s1 = (float)spa[hrow * W_IMG + wc1];

#pragma unroll
    for (int am = 0; am < 2; ++am) {
        const v8f accL = am ? acc10 : acc00;   // N tile nt
        const v8f accR = am ? acc11 : acc01;   // N tile nt+1
#pragma unroll
        for (int r = 0; r < 8; ++r) {
            const int co = coB + (mi2 + am) * 16 + lhalf * 8 + r;
            const float c0v = chm[2 * co];
            const float c1v = chm[2 * co + 1];
            const float mx  = fmaxf(c0v, c1v);
            const float e0  = __expf(c0v - mx);
            const float e1  = __expf(c1v - mx);
            const float p0  = e0 / (e0 + e1);
            const float dg  = rintf(p0);          // dense gate
            const float sg  = rintf(1.0f - p0);   // sparse gate
            const size_t rowbase = (size_t)co * HW + (size_t)hrow * W_IMG;
            out[rowbase + wc0] = fmaxf(accL[r] * (dg + sg * s0), 0.0f);
            out[rowbase + wc1] = fmaxf(accR[r] * (dg + sg * s1), 0.0f);
        }
    }
}

// Second output: ch_round = round(softmax(ch_mask, axis=2)), 256 x 2 floats.
__global__ void chround_kernel(const float* __restrict__ chm, float* __restrict__ out)
{
    const int c = blockIdx.x * blockDim.x + threadIdx.x;
    if (c < C_CH) {
        const float a0 = chm[2 * c];
        const float a1 = chm[2 * c + 1];
        const float mx = fmaxf(a0, a1);
        const float e0 = __expf(a0 - mx);
        const float e1 = __expf(a1 - mx);
        const float p0 = e0 / (e0 + e1);
        out[2 * c]     = rintf(p0);
        out[2 * c + 1] = rintf(1.0f - p0);
    }
}

extern "C" void kernel_launch(void* const* d_in, const int* in_sizes, int n_in,
                              void* d_out, int out_size, void* d_ws, size_t ws_size,
                              hipStream_t stream) {
    (void)in_sizes; (void)n_in; (void)out_size; (void)ws_size;
    const float* x   = (const float*)d_in[0];   // [1,256,384,384] f32
    const int*   spa = (const int*)d_in[1];     // [1,1,384,384] i32
    const float* wgt = (const float*)d_in[2];   // [256,256,3,3] f32
    const float* chm = (const float*)d_in[3];   // [1,256,2] f32
    float* out = (float*)d_out;                 // [256*384*384] conv out, then [512] ch_round
    unsigned short* Wt = (unsigned short*)d_ws; // bf16 Wt[9][256][256] = 1.18 MB scratch

    // one-time weight bf16 transpose (deterministic, re-run every call)
    prep_weights<<<(9 * C_CH * C_CH) / 256, 256, 0, stream>>>(wgt, Wt);

    chround_kernel<<<1, 256, 0, stream>>>(chm, out + (size_t)C_CH * HW);

    // grid: 2304 spatial blocks (384 rows x 6 segments of 64 px) x 2 channel blocks of 128
    masked_conv3x3_wmma_bf16<<<dim3(2304, 2), 256, 0, stream>>>(x, spa, Wt, chm, out);
}
